// Decoder_46875273069358
// MI455X (gfx1250) — matrix operand order repaired and checked by reading, not compiled, not run on hardware
//
#include <hip/hip_runtime.h>
#include <math.h>

// Problem constants (from reference): H=1024 hidden, B=256 batch, T=512 steps.
#define Hh   1024
#define Bsz  256
#define Tsz  512
#define NWG  64        // 4 B-blocks x 16 J-blocks
#define SW   1032      // padded LDS row stride (halfwords): 2064B, odd multiple of 16B
#define SOW  64        // h' output slab row stride (halfwords): contiguous for TDM store

typedef __attribute__((ext_vector_type(16))) __bf16        v16bf;
typedef __attribute__((ext_vector_type(8)))  float         v8f;
typedef __attribute__((ext_vector_type(4)))  unsigned int  u32x4;
typedef __attribute__((ext_vector_type(8)))  unsigned int  u32x8;

union Frag { v16bf bf; u32x4 q[2]; };
union F4   { u32x4 q;  float f[4]; };

__device__ __forceinline__ unsigned short f2bf(float f) {
    unsigned int u = __builtin_bit_cast(unsigned int, f);
    unsigned int r = u + 0x7fffu + ((u >> 16) & 1u);   // round-to-nearest-even
    return (unsigned short)(r >> 16);
}

// Branchless tanh: 1 - 2/(e^{2x}+1), via hardware v_exp_f32 / v_rcp_f32.
__device__ __forceinline__ float fast_tanh(float x) {
    float e2x = __builtin_amdgcn_exp2f(x * 2.8853900817779268f);  // 2*log2(e)
    return 1.0f - 2.0f * __builtin_amdgcn_rcpf(e2x + 1.0f);
}

// ---- CDNA5 async global->LDS copy (ASYNCcnt-tracked) via inline asm.
// The LDS-address VGPR wants the wave-relative LDS byte offset; a generic
// pointer to LDS is {aperture_hi32, lds_offset32}, so truncation yields it.
__device__ __forceinline__ void async_g2l_b128(const unsigned short* gsrc,
                                               unsigned short* ldst) {
    unsigned int loff = (unsigned int)(unsigned long long)ldst;
    asm volatile("global_load_async_to_lds_b128 %0, %1, off"
                 :
                 : "v"(loff), "v"(gsrc)
                 : "memory");
}

__device__ __forceinline__ void wait_async0() {
    asm volatile("s_wait_asynccnt 0x0" ::: "memory");
}

__device__ __forceinline__ void wait_tensor0() {
#if __has_builtin(__builtin_amdgcn_s_wait_tensorcnt)
    __builtin_amdgcn_s_wait_tensorcnt(0);
#else
    asm volatile("s_wait_tensorcnt 0x0" ::: "memory");
#endif
}

// ---------------------------------------------------------------------------
// Init kernel: zero the global-barrier counter, seed y with b_out, convert the
// initial hidden state f32 -> bf16 into ping-pong buffer 0. Stream ordering
// makes this a deterministic reset every kernel_launch call (graph-safe).
// ---------------------------------------------------------------------------
__global__ void rnn_init(const float* __restrict__ hidden,
                         const float* __restrict__ b_out,
                         float* __restrict__ y,
                         unsigned short* __restrict__ hbuf0,
                         unsigned int* __restrict__ cnt) {
    int idx = blockIdx.x * blockDim.x + threadIdx.x;
    if (idx == 0) *cnt = 0u;
    if (idx < Tsz * Bsz) y[idx] = b_out[0];
    if (idx < Bsz * Hh)  hbuf0[idx] = f2bf(hidden[idx]);
}

// ---------------------------------------------------------------------------
// Persistent RNN kernel. 64 WGs x 256 threads (8 wave32).
//   WG (wg&3)   -> B-block of 64 rows;  WG (wg>>2) -> J-block of 64 units
//   wave (w&3)  -> 16-row B tile;       wave (w>>2) -> pair of 16-col J tiles
// W_hh slice resident in LDS (bf16) for all 512 steps; h staged into LDS each
// step with async loads; h' scattered into an LDS slab and written back with a
// single Tensor-Data-Mover store per WG; one device-wide barrier per step.
// ---------------------------------------------------------------------------
__global__ __launch_bounds__(256, 1)
void rnn_persistent(const float* __restrict__ init_in,
                    const float* __restrict__ targets,
                    const float* __restrict__ W_ih,
                    const float* __restrict__ W_hh,
                    const float* __restrict__ b_ih,
                    const float* __restrict__ b_hh,
                    const float* __restrict__ W_out,
                    float* __restrict__ y,
                    unsigned short* __restrict__ hbuf,
                    unsigned int* __restrict__ cnt)
{
    extern __shared__ unsigned short smem[];
    unsigned short* sW = smem;             // [64][SW]  bf16 W_hh slice (rows = j)
    unsigned short* sH = smem + 64 * SW;   // [64][SW]  bf16 h slice   (rows = b)
    unsigned short* sO = sH;               // [64][SOW] h' slab (reuses sH region)

    const int tid   = threadIdx.x;
    const int wg    = blockIdx.x;
    const int bwg   = (wg & 3) * 64;       // global B base of this WG
    const int jwg   = (wg >> 2) * 64;      // global J base of this WG
    const int warp  = tid >> 5;
    const int lane  = tid & 31;
    const int half  = lane >> 4;
    const int l     = lane & 15;
    const int brow0 = (warp & 3) * 16;     // B tile base within WG block
    const int jrel0 = (warp >> 2) * 32;    // J tile-pair base within WG block

    // ---- one-time: W_hh slice -> LDS (f32 -> bf16), resident for all steps
    for (int i = tid; i < 64 * 1024; i += 256) {
        int jr = i >> 10, k = i & 1023;
        sW[jr * SW + k] = f2bf(W_hh[(size_t)(jwg + jr) * Hh + k]);
    }

    // ---- per-lane constants: each lane owns two j columns (one per tile)
    const int jg0 = jwg + jrel0 + l;
    const int jg1 = jg0 + 16;
    const float wih0 = W_ih[jg0], wih1 = W_ih[jg1];
    const float bia0 = b_ih[jg0] + b_hh[jg0];
    const float bia1 = b_ih[jg1] + b_hh[jg1];
    const float wo0  = W_out[jg0], wo1 = W_out[jg1];

    // ---- TDM descriptor pieces that are step-invariant
    const unsigned int lds_sO =
        __builtin_amdgcn_readfirstlane((unsigned int)(unsigned long long)sO);

    __syncthreads();

    for (int t = 0; t < Tsz; ++t) {
        const unsigned short* hc = hbuf + (size_t)(t & 1) * Bsz * Hh;
        unsigned short*       hn = hbuf + (size_t)((t + 1) & 1) * Bsz * Hh;

        // ---- stage this WG's 64 rows of h (bf16) into LDS, async b128 copies
        for (int i = tid; i < 64 * 128; i += 256) {      // 128 x 16B per row
            int r = i >> 7, c = i & 127;
            async_g2l_b128(hc + (size_t)(bwg + r) * Hh + c * 8,
                           sH + r * SW + c * 8);
        }
        wait_async0();
        __syncthreads();

        // ---- GEMM: two 16x16 f32 accumulators, K = 1024 in chunks of 32
        v8f c0 = {}; v8f c1 = {};
        #pragma unroll 4
        for (int kc = 0; kc < 32; ++kc) {
            // A fragment (16x32): lane<16 -> K 0-15; lane>=16 -> K 16-31
            const int ka = kc * 32 + half * 16;
            const unsigned short* ap = sH + (brow0 + l) * SW + ka;
            Frag A;
            A.q[0] = *(const u32x4*)(ap);
            A.q[1] = *(const u32x4*)(ap + 8);
            // B fragments (32x16): lane<16 -> K 0-15 (col=l); lane>=16 -> K 16-31
            const int kb = kc * 32 + half * 16;
            const unsigned short* bp0 = sW + (jrel0 + l) * SW + kb;
            Frag B0, B1;
            B0.q[0] = *(const u32x4*)(bp0);
            B0.q[1] = *(const u32x4*)(bp0 + 8);
            const unsigned short* bp1 = bp0 + 16 * SW;
            B1.q[0] = *(const u32x4*)(bp1);
            B1.q[1] = *(const u32x4*)(bp1 + 8);

            c0 = __builtin_amdgcn_wmma_f32_16x16x32_bf16(false, A.bf, false, B0.bf,
                                                         (short)0, c0, false, false);
            c1 = __builtin_amdgcn_wmma_f32_16x16x32_bf16(false, A.bf, false, B1.bf,
                                                         (short)0, c1, false, false);
        }

        __syncthreads();   // all waves done reading sH; safe to reuse as sO

        // ---- epilogue: + x*W_ih + biases, tanh, y readout, h' -> LDS slab
        const float* xsrc = (t == 0) ? init_in : (targets + (size_t)(t - 1) * Bsz);
        const int bbase = bwg + brow0 + half * 8;   // multiple of 8 -> 32B aligned
        F4 x0, x1;
        x0.q = *(const u32x4*)(xsrc + bbase);
        x1.q = *(const u32x4*)(xsrc + bbase + 4);

        const int orow0 = brow0 + half * 8;         // row within WG's 64-row slab
        #pragma unroll
        for (int r = 0; r < 8; ++r) {
            const float x  = (r < 4) ? x0.f[r] : x1.f[r - 4];
            const float h0 = fast_tanh(c0[r] + x * wih0 + bia0);
            const float h1 = fast_tanh(c1[r] + x * wih1 + bia1);
            sO[(orow0 + r) * SOW + jrel0 + l]      = f2bf(h0);
            sO[(orow0 + r) * SOW + jrel0 + l + 16] = f2bf(h1);
            // y[t][b] += sum_j h'[b][j] * W_out[j] (16-lane tree, then atomic)
            float p = h0 * wo0 + h1 * wo1;
            p += __shfl_xor(p, 1);
            p += __shfl_xor(p, 2);
            p += __shfl_xor(p, 4);
            p += __shfl_xor(p, 8);
            if (l == 0) atomicAdd(&y[(size_t)t * Bsz + (bbase + r)], p);
        }

        __syncthreads();   // h' slab complete

        // ---- Tensor Data Mover: one DMA store of the 64x64 bf16 slab -> hn.
        // 2D descriptor: tile_dim0=64, tile_dim1=64, data_size=2B,
        // tensor_dim0_stride=1024 elements; LDS side read contiguously.
        if (warp == 0) {
            const unsigned long long ga =
                (unsigned long long)hn + 2ull * ((size_t)bwg * Hh + jwg);
            u32x4 g0;
            g0[0] = 1u;                                   // count=1, user descriptor
            g0[1] = lds_sO;                               // lds_addr (bytes)
            g0[2] = (unsigned int)ga;                     // global_addr[31:0]
            g0[3] = ((unsigned int)(ga >> 32) & 0x01FFFFFFu)  // global_addr[56:32]
                    | (2u << 30);                         // type = 2 ("image")
            u32x8 g1;
            g1[0] = 0x00010000u;    // wg_mask=0, data_size=1 (2 bytes), no pad/iter
            g1[1] = 1024u << 16;    // tensor_dim0 = 1024 (bits 79:48, low half)
            g1[2] = 256u  << 16;    // tensor_dim0 hi=0 | tensor_dim1 = 256 (low half)
            g1[3] = 64u   << 16;    // tensor_dim1 hi=0 | tile_dim0 = 64
            g1[4] = 64u;            // tile_dim1 = 64 | tile_dim2 = 0
            g1[5] = 1024u;          // tensor_dim0_stride = 1024 (low 32)
            g1[6] = 0u;             // stride hi | tensor_dim1_stride lo
            g1[7] = 0u;
            asm volatile("tensor_store_from_lds %0, %1"
                         :
                         : "s"(g0), "s"(g1)
                         : "memory");
            wait_tensor0();
        }

        // ---- device-wide step barrier (h' must be globally visible)
        __threadfence();
        __syncthreads();
        if (tid == 0) {
            __hip_atomic_fetch_add(cnt, 1u, __ATOMIC_RELEASE, __HIP_MEMORY_SCOPE_AGENT);
            const unsigned int target = (unsigned int)(t + 1) * NWG;
            while (__hip_atomic_load(cnt, __ATOMIC_ACQUIRE, __HIP_MEMORY_SCOPE_AGENT) < target) {
                __builtin_amdgcn_s_sleep(8);
            }
        }
        __syncthreads();
    }
}

// ---------------------------------------------------------------------------
extern "C" void kernel_launch(void* const* d_in, const int* in_sizes, int n_in,
                              void* d_out, int out_size, void* d_ws, size_t ws_size,
                              hipStream_t stream) {
    const float* init_in = (const float*)d_in[0];   // [B,1]
    const float* hidden  = (const float*)d_in[1];   // [B,H]
    const float* targets = (const float*)d_in[2];   // [T,B,1]
    const float* W_ih    = (const float*)d_in[3];   // [H,1]
    const float* W_hh    = (const float*)d_in[4];   // [H,H]
    const float* b_ih    = (const float*)d_in[5];   // [H]
    const float* b_hh    = (const float*)d_in[6];   // [H]
    const float* W_out   = (const float*)d_in[7];   // [1,H]
    const float* b_out   = (const float*)d_in[8];   // [1]
    (void)in_sizes; (void)n_in; (void)out_size; (void)ws_size;

    float* y = (float*)d_out;                       // [T,B,1] f32

    // workspace: [0..3] barrier counter, then bf16 h ping-pong buffers (1 MiB)
    unsigned int*   cnt  = (unsigned int*)d_ws;
    unsigned short* hbuf = (unsigned short*)((char*)d_ws + 256);

    rnn_init<<<(Bsz * Hh + 255) / 256, 256, 0, stream>>>(hidden, b_out, y, hbuf, cnt);

    const size_t lds_bytes = (size_t)128 * SW * sizeof(unsigned short);  // 264192 B < 320 KB
    rnn_persistent<<<NWG, 256, lds_bytes, stream>>>(init_in, targets, W_ih, W_hh,
                                                    b_ih, b_hh, W_out, y, hbuf, cnt);
}
